// VQ_70858370449571
// MI455X (gfx1250) — matrix operand order repaired and checked by reading, not compiled, not run on hardware
//
#include <hip/hip_runtime.h>
#include <hip/hip_bf16.h>

typedef __attribute__((ext_vector_type(16))) _Float16 v16h;
typedef __attribute__((ext_vector_type(8)))  _Float16 v8h;
typedef __attribute__((ext_vector_type(4)))  _Float16 v4h;
typedef __attribute__((ext_vector_type(8)))  float    v8f;

#define K_CODES 512
#define DIM     64
#define ROWS_PER_BLOCK 128   // 8 waves x 16 rows

// ---------------------------------------------------------------------------
// Kernel 1: precompute f16 codebook + squared norms into workspace.
// cbh: [512*64] _Float16, cnorm: [512] float
// ---------------------------------------------------------------------------
__global__ void vq_prep_kernel(const float* __restrict__ cb,
                               _Float16* __restrict__ cbh,
                               float* __restrict__ cnorm) {
    int t = blockIdx.x * blockDim.x + threadIdx.x;
    if (t >= K_CODES) return;
    const float4* src = (const float4*)(cb + (size_t)t * DIM);
    float acc = 0.0f;
#pragma unroll
    for (int i = 0; i < 16; ++i) {
        float4 v = src[i];
        acc = __builtin_fmaf(v.x, v.x, acc);
        acc = __builtin_fmaf(v.y, v.y, acc);
        acc = __builtin_fmaf(v.z, v.z, acc);
        acc = __builtin_fmaf(v.w, v.w, acc);
        v4h h;
        h.x = (_Float16)v.x; h.y = (_Float16)v.y;
        h.z = (_Float16)v.z; h.w = (_Float16)v.w;
        *(v4h*)(cbh + (size_t)t * DIM + 4 * i) = h;
    }
    cnorm[t] = acc;
}

// ---------------------------------------------------------------------------
// Kernel 2: main VQ. One block = 8 waves = 128 rows. LDS holds the entire f16
// codebook (64KB) + norms (2KB). Each wave: 16 rows x 512 codes via
// v_wmma_f32_16x16x32_f16 (2 wmma per 16-code tile, K=64).
// ---------------------------------------------------------------------------
#define PACK8(dstv, off, f0, f1)                                             \
    dstv[off + 0] = (_Float16)f0.x; dstv[off + 1] = (_Float16)f0.y;          \
    dstv[off + 2] = (_Float16)f0.z; dstv[off + 3] = (_Float16)f0.w;          \
    dstv[off + 4] = (_Float16)f1.x; dstv[off + 5] = (_Float16)f1.y;          \
    dstv[off + 6] = (_Float16)f1.z; dstv[off + 7] = (_Float16)f1.w;

__global__ __launch_bounds__(256) void vq_main_kernel(
        const float*    __restrict__ x,
        const float*    __restrict__ cb,
        const _Float16* __restrict__ cbh_g,
        const float*    __restrict__ cnorm_g,
        float*          __restrict__ out,
        int nrows) {
    extern __shared__ char smem[];
    _Float16* cbh = (_Float16*)smem;                             // 64 KB
    float*    cns = (float*)(smem + K_CODES * DIM * sizeof(_Float16)); // 2 KB

    // ---- cooperative LDS fill: 64KB f16 codebook + 512 norms ----
    {
        const uint4* src = (const uint4*)cbh_g;
        uint4*       dst = (uint4*)cbh;
        int t = threadIdx.x;
#pragma unroll
        for (int i = 0; i < 16; ++i)
            dst[t + 256 * i] = src[t + 256 * i];
        cns[t]       = cnorm_g[t];
        cns[t + 256] = cnorm_g[t + 256];
    }
    __syncthreads();

    const int wave = threadIdx.x >> 5;
    const int lane = threadIdx.x & 31;
    const int hh   = lane >> 4;          // half-wave: 0 or 1
    const int l16  = lane & 15;
    const int row0 = blockIdx.x * ROWS_PER_BLOCK + wave * 16;
    if (row0 >= nrows) return;
    const int m = row0 + l16;            // A-matrix row owned by this lane

    // ---- A fragments: 16x32 f16, ISA layout ----
    // lanes 0-15: K = k0..k0+7 then k0+16..k0+23 with k0=0
    // lanes 16-31: same with k0=8
    const float4* xr  = (const float4*)(x + (size_t)m * DIM);
    const int     k0q = hh * 2;          // k0 in float4 units
    float4 p0 = xr[k0q + 0], p1 = xr[k0q + 1];
    float4 p2 = xr[k0q + 4], p3 = xr[k0q + 5];
    float4 q0 = xr[k0q + 8], q1 = xr[k0q + 9];
    float4 q2 = xr[k0q + 12], q3 = xr[k0q + 13];
    v16h a0, a1;
    PACK8(a0, 0, p0, p1); PACK8(a0, 8, p2, p3);   // K chunk 0..31
    PACK8(a1, 0, q0, q1); PACK8(a1, 8, q2, q3);   // K chunk 32..63

    // ---- running argmin over code tiles ----
    float best[8];
    int   bidx[8];
#pragma unroll
    for (int r = 0; r < 8; ++r) { best[r] = 3.4e38f; bidx[r] = 0; }

    const int kb0 = hh * 8;              // B layout: lanes 0-15 K 0-7,16-23; 16-31 K 8-15,24-31

#pragma unroll 4
    for (int t = 0; t < K_CODES / 16; ++t) {
        const int n = t * 16 + l16;      // code column owned by this lane
        const _Float16* bp = cbh + (size_t)n * DIM + kb0;
        v8h b0l = *(const v8h*)(bp);
        v8h b0h = *(const v8h*)(bp + 16);
        v8h b1l = *(const v8h*)(bp + 32);
        v8h b1h = *(const v8h*)(bp + 48);
        v16h b0 = __builtin_shufflevector(b0l, b0h, 0,1,2,3,4,5,6,7,8,9,10,11,12,13,14,15);
        v16h b1 = __builtin_shufflevector(b1l, b1h, 0,1,2,3,4,5,6,7,8,9,10,11,12,13,14,15);

        v8f acc = {};
        acc = __builtin_amdgcn_wmma_f32_16x16x32_f16(false, a0, false, b0,
                                                     (short)0, acc, false, false);
        acc = __builtin_amdgcn_wmma_f32_16x16x32_f16(false, a1, false, b1,
                                                     (short)0, acc, false, false);
        const float cn = cns[n];
#pragma unroll
        for (int r = 0; r < 8; ++r) {
            float d = __builtin_fmaf(-2.0f, acc[r], cn);  // ||c||^2 - 2 x.c
            if (d < best[r]) { best[r] = d; bidx[r] = n; }
        }
    }

    // ---- reduce (min, idx) across the 16 N-columns (stay within half-wave) ----
#pragma unroll
    for (int r = 0; r < 8; ++r) {
        float v = best[r];
        int   i = bidx[r];
#pragma unroll
        for (int off = 1; off < 16; off <<= 1) {
            float v2 = __shfl_xor(v, off, 32);
            int   i2 = __shfl_xor(i, off, 32);
            if (v2 < v || (v2 == v && i2 < i)) { v = v2; i = i2; }
        }
        bidx[r] = i;
    }

    // ---- gather fp32 codebook rows, coalesced float4 writes ----
    // C/D layout: VGPR r -> row (row0 + r) for lanes 0-15, (row0 + 8 + r) for 16-31
    const float4* cbv  = (const float4*)cb;
    float4*       outv = (float4*)out;
#pragma unroll
    for (int r = 0; r < 8; ++r) {
        const int row = row0 + hh * 8 + r;
        const int idx = bidx[r];
        outv[(size_t)row * 16 + l16] = cbv[(size_t)idx * 16 + l16];
    }
}

// ---------------------------------------------------------------------------
extern "C" void kernel_launch(void* const* d_in, const int* in_sizes, int n_in,
                              void* d_out, int out_size, void* d_ws, size_t ws_size,
                              hipStream_t stream) {
    const float* x  = (const float*)d_in[0];   // [32,64,64,64] fp32
    const float* cb = (const float*)d_in[1];   // [512,64] fp32
    float* out = (float*)d_out;

    const int nrows = in_sizes[0] / DIM;       // 131072

    _Float16* cbh   = (_Float16*)d_ws;                                   // 64 KB
    float*    cnorm = (float*)((char*)d_ws + K_CODES * DIM * sizeof(_Float16)); // 2 KB

    vq_prep_kernel<<<2, 256, 0, stream>>>(cb, cbh, cnorm);

    const size_t shmem = K_CODES * DIM * sizeof(_Float16) + K_CODES * sizeof(float);
    const int grid = (nrows + ROWS_PER_BLOCK - 1) / ROWS_PER_BLOCK;      // 1024
    vq_main_kernel<<<grid, 256, shmem, stream>>>(x, cb, cbh, cnorm, out, nrows);
}
